// DDH_54262616817680
// MI455X (gfx1250) — matrix operand order repaired and checked by reading, not compiled, not run on hardware
//
#include <hip/hip_runtime.h>
#include <hip/hip_bf16.h>

// ---------------------------------------------------------------------------
// CDNA5 / gfx1250 CenterNet-style DCN head.
// All contractions run as WMMA bf16 GEMMs (v_wmma_f32_16x16x32_bf16):
//  - weights pre-converted to bf16 matrices, staged to LDS with
//    global_load_async_to_lds_b128 (ASYNCcnt) into double-buffered tiles
//  - activations f32 in HBM, converted to bf16 while staging B tiles
//  - fused bias / BN / ReLU / scale epilogue, NCHW f32 stores
// ---------------------------------------------------------------------------

typedef __attribute__((ext_vector_type(16))) __bf16 v16bf;
typedef __attribute__((ext_vector_type(8)))  float  v8f;

#define TILE_S 128
#define TILE_O 64
#define TILE_K 32
#define A_STRIDE 40   // bf16 units: 80B rows -> 16B aligned (b128), 20-dword bank skew
#define B_STRIDE 36   // bf16 units: 72B rows ->  4B aligned (pair loads), 18-dword skew

__device__ __forceinline__ unsigned short f2bf(float f) {
    union { float f; unsigned u; } v; v.f = f;
    unsigned r = v.u + 0x7FFFu + ((v.u >> 16) & 1u);   // round-to-nearest-even
    return (unsigned short)(r >> 16);
}

union FragB { v16bf v; unsigned u[8]; };
union FragC { v8f   v; float    f[8]; };

// One per-thread 16B async global->LDS copy (wave-level ASYNCcnt tracking).
__device__ __forceinline__ void async_b128(const unsigned short* g, unsigned short* l) {
    const unsigned lofs = (unsigned)(unsigned long long)l;   // LDS byte offset = addr[31:0]
    asm volatile("global_load_async_to_lds_b128 %0, %1, off"
                 :: "v"(lofs), "v"(g)
                 : "memory");
}
__device__ __forceinline__ void wait_async0() {
    asm volatile("s_wait_asynccnt 0x0" ::: "memory");
}

// GEMM: Out[o][s] = sum_k A[o][k] * B[k][s], fused epilogue, NCHW f32 store.
// A: bf16 weights [Opad][K] (Opad = ceil(O/64)*64), K multiple of 32.
// MODE 0: B = precomputed bf16 matrix [K][S]      (DCN einsum)
// MODE 1: B = implicit im2col of 3x3 pad-1 conv   (k = c*9 + ky*3 + kx)
// MODE 2: B = 1x1 conv (k = c)
// flags: 1=bias, 2=batchnorm, 4=relu, 8=scale by outscale
template <int MODE>
__global__ void __launch_bounds__(256)
gemm_conv_wmma(const unsigned short* __restrict__ Aw,
               const float* __restrict__ Xin,
               const unsigned short* __restrict__ Bmat,
               float* __restrict__ Out,
               const float* __restrict__ bias,
               const float* __restrict__ gamma,
               const float* __restrict__ beta,
               const float* __restrict__ mean,
               const float* __restrict__ var,
               int O, int K, int Bn, int C, int H, int W,
               int flags, float outscale)
{
    const int HW = H * W;
    const int S  = Bn * HW;
    const int s_base = blockIdx.x * TILE_S;
    const int o_base = blockIdx.y * TILE_O;

    __shared__ unsigned short As[2][TILE_O][A_STRIDE];   // [o][k]
    __shared__ unsigned short Bs[2][TILE_S][B_STRIDE];   // [s][k]  (K contiguous)

    const int tid  = threadIdx.x;
    const int lane = tid & 31;
    const int wave = tid >> 5;
    const int wo   = wave & 1;    // O half (0..1) -> 32 rows
    const int wsx  = wave >> 1;   // S quarter (0..3) -> 32 cols

    v8f acc00 = {}, acc01 = {}, acc10 = {}, acc11 = {};

    // A staging: 64x32 bf16 tile = 256 threads x one b128 (8 bf16)
    const int a_row = tid >> 2;
    const int a_seg = (tid & 3) * 8;
    // B staging: 32(k) x 128(s), thread -> k row tid>>3, 16 consecutive s
    const int b_k  = tid >> 3;
    const int b_s0 = (tid & 7) * 16;

    auto stage = [&](int buf, int kb) {
        // ---- A tile: async b128 from bf16 weight matrix ----
        async_b128(Aw + (size_t)(o_base + a_row) * K + kb + a_seg,
                   &As[buf][a_row][a_seg]);
        // ---- B tile ----
        if (MODE == 0) {
            const int kg = kb + b_k;
            const unsigned short* src = Bmat + (size_t)kg * S + s_base + b_s0;
            if (kb + TILE_K < K)
                __builtin_prefetch(src + (size_t)TILE_K * S, 0, 1);
#pragma unroll 4
            for (int j = 0; j < 16; ++j) {
                const int sg = s_base + b_s0 + j;
                Bs[buf][b_s0 + j][b_k] = (sg < S) ? src[j] : (unsigned short)0;
            }
        } else if (MODE == 1) {
            const int kg = kb + b_k;
            const int c  = kg / 9;
            const int r  = kg - c * 9;
            const int ky = r / 3 - 1;
            const int kx = r - (r / 3) * 3 - 1;
#pragma unroll 4
            for (int j = 0; j < 16; ++j) {
                const int sg = s_base + b_s0 + j;
                float f = 0.0f;
                if (sg < S) {
                    const int b  = sg / HW;
                    const int hw = sg - b * HW;
                    const int h  = hw / W;
                    const int w  = hw - h * W;
                    const int y  = h + ky;
                    const int x  = w + kx;
                    if (y >= 0 && y < H && x >= 0 && x < W)
                        f = Xin[(size_t)(b * C + c) * HW + y * W + x];
                }
                Bs[buf][b_s0 + j][b_k] = f2bf(f);
            }
        } else { // MODE 2: 1x1
            const int c = kb + b_k;
#pragma unroll 4
            for (int j = 0; j < 16; ++j) {
                const int sg = s_base + b_s0 + j;
                float f = 0.0f;
                if (sg < S) {
                    const int b  = sg / HW;
                    const int hw = sg - b * HW;
                    f = Xin[(size_t)(b * C + c) * HW + hw];
                }
                Bs[buf][b_s0 + j][b_k] = f2bf(f);
            }
        }
    };

    int buf = 0;
    stage(0, 0);
    for (int kb = 0; kb < K; kb += TILE_K) {
        wait_async0();        // own async A writes done
        __syncthreads();      // all waves' tile[buf] writes visible
        const int nb = buf ^ 1;
        if (kb + TILE_K < K)
            stage(nb, kb + TILE_K);   // overlap next-tile staging with compute

        // ---- build fragments: A and B both use k = 16*(lane>>4) + s ----
        const int khalf = (lane >> 4) * 16;           // A: 0 or 16
        const int row0  = wo * 32 + (lane & 15);
        FragB a0, a1;
        {
            const unsigned* p0 = (const unsigned*)&As[buf][row0][0];
            const unsigned* p1 = (const unsigned*)&As[buf][row0 + 16][0];
#pragma unroll
            for (int i = 0; i < 4; ++i) {
                a0.u[i]     = p0[(khalf >> 1) + i];
                a0.u[4 + i] = p0[((khalf + 8) >> 1) + i];
                a1.u[i]     = p1[(khalf >> 1) + i];
                a1.u[4 + i] = p1[((khalf + 8) >> 1) + i];
            }
        }
        FragB b0, b1;
        {
            const int scol = wsx * 32 + (lane & 15);
            const int koff = (lane >> 4) * 16;        // B: 0 or 16
            const unsigned* p0 = (const unsigned*)&Bs[buf][scol][0];
            const unsigned* p1 = (const unsigned*)&Bs[buf][scol + 16][0];
#pragma unroll
            for (int i = 0; i < 8; ++i) {
                b0.u[i] = p0[(koff >> 1) + i];
                b1.u[i] = p1[(koff >> 1) + i];
            }
        }
        acc00 = __builtin_amdgcn_wmma_f32_16x16x32_bf16(false, a0.v, false, b0.v, (short)0, acc00, false, false);
        acc01 = __builtin_amdgcn_wmma_f32_16x16x32_bf16(false, a0.v, false, b1.v, (short)0, acc01, false, false);
        acc10 = __builtin_amdgcn_wmma_f32_16x16x32_bf16(false, a1.v, false, b0.v, (short)0, acc10, false, false);
        acc11 = __builtin_amdgcn_wmma_f32_16x16x32_bf16(false, a1.v, false, b1.v, (short)0, acc11, false, false);
        buf = nb;
    }

    // ---- epilogue: D layout -> col = lane&15, rows = (lane>>4)*8 + r ----
    const bool has_bias  = flags & 1;
    const bool has_bn    = flags & 2;
    const bool has_relu  = flags & 4;
    const bool has_scale = flags & 8;

    FragC cc[2][2];
    cc[0][0].v = acc00; cc[0][1].v = acc01;
    cc[1][0].v = acc10; cc[1][1].v = acc11;

#pragma unroll
    for (int i = 0; i < 2; ++i) {
        const int orow0 = o_base + wo * 32 + i * 16 + (lane >> 4) * 8;
#pragma unroll
        for (int r = 0; r < 8; ++r) {
            const int o = orow0 + r;
            if (o >= O) continue;
            const float bv = has_bias ? bias[o] : 0.0f;
            float sc = 1.0f, mn = 0.0f, bt = 0.0f;
            if (has_bn) {
                sc = gamma[o] * rsqrtf(var[o] + 1e-5f);
                mn = mean[o];
                bt = beta[o];
            }
#pragma unroll
            for (int j = 0; j < 2; ++j) {
                const int s = s_base + wsx * 32 + j * 16 + (lane & 15);
                if (s >= S) continue;
                float v = cc[i][j].f[r] + bv;
                if (has_bn)    v = (v - mn) * sc + bt;
                if (has_relu)  v = fmaxf(v, 0.0f);
                if (has_scale) v *= outscale;
                const int b  = s / HW;
                const int hw = s - b * HW;
                Out[(size_t)(b * O + o) * HW + hw] = v;
            }
        }
    }
}

// f32 -> bf16 weight pre-conversion
__global__ void __launch_bounds__(256)
f32_to_bf16_kernel(const float* __restrict__ src, unsigned short* __restrict__ dst, int n)
{
    const int i = blockIdx.x * blockDim.x + threadIdx.x;
    if (i < n) dst[i] = f2bf(src[i]);
}

// Deformable bilinear gather: writes V[(c*9+kk)][b*HW+hw] (bf16, mask fused)
__global__ void __launch_bounds__(256)
dcn_sample(const float* __restrict__ X, const float* __restrict__ Om,
           unsigned short* __restrict__ Vmat, int Bn, int C, int H, int W)
{
    const int HW = H * W;
    const int S  = Bn * HW;
    const int idx = blockIdx.x * blockDim.x + threadIdx.x;
    const int total = Bn * 9 * HW;
    if (idx >= total) return;
    const int b  = idx / (9 * HW);
    const int r  = idx - b * 9 * HW;
    const int kk = r / HW;
    const int hw = r - kk * HW;
    const int h  = hw / W;
    const int w  = hw - h * W;

    const float* omb = Om + (size_t)b * 27 * HW;
    const float dy   = omb[(2 * kk) * HW + hw];
    const float dx   = omb[(2 * kk + 1) * HW + hw];
    const float mval = 1.0f / (1.0f + __expf(-omb[(18 + kk) * HW + hw]));

    const float py = dy + (float)h + (float)(kk / 3 - 1);
    const float px = dx + (float)w + (float)(kk % 3 - 1);
    const float y0f = floorf(py), x0f = floorf(px);
    const float fy = py - y0f, fx = px - x0f;
    const int y0 = (int)y0f, x0 = (int)x0f;
    const int y1 = y0 + 1, x1 = x0 + 1;
    const float vy0 = (y0 >= 0 && y0 < H) ? 1.0f : 0.0f;
    const float vy1 = (y1 >= 0 && y1 < H) ? 1.0f : 0.0f;
    const float vx0 = (x0 >= 0 && x0 < W) ? 1.0f : 0.0f;
    const float vx1 = (x1 >= 0 && x1 < W) ? 1.0f : 0.0f;
    const int cy0 = min(max(y0, 0), H - 1), cy1 = min(max(y1, 0), H - 1);
    const int cx0 = min(max(x0, 0), W - 1), cx1 = min(max(x1, 0), W - 1);
    const float w00 = (1.0f - fy) * (1.0f - fx) * vy0 * vx0;
    const float w01 = (1.0f - fy) * fx          * vy0 * vx1;
    const float w10 = fy          * (1.0f - fx) * vy1 * vx0;
    const float w11 = fy          * fx          * vy1 * vx1;
    const int i00 = cy0 * W + cx0, i01 = cy0 * W + cx1;
    const int i10 = cy1 * W + cx0, i11 = cy1 * W + cx1;

    const float* xb  = X + (size_t)b * C * HW;
    const size_t col = (size_t)b * HW + hw;
    for (int c = 0; c < C; ++c) {
        const float* xc = xb + (size_t)c * HW;
        const float v = w00 * xc[i00] + w01 * xc[i01] + w10 * xc[i10] + w11 * xc[i11];
        Vmat[(size_t)(c * 9 + kk) * S + col] = f2bf(v * mval);
    }
}

// 2x bilinear upsample (align_corners) + optional residual add
__global__ void __launch_bounds__(256)
upsample2x_add(const float* __restrict__ In, const float* __restrict__ Res,
               float* __restrict__ Out, int Bn, int C, int H, int W)
{
    const int OH = 2 * H, OW = 2 * W;
    const long long total = (long long)Bn * C * OH * OW;
    const long long idx = (long long)blockIdx.x * blockDim.x + threadIdx.x;
    if (idx >= total) return;
    const int ow = (int)(idx % OW);
    const long long t1 = idx / OW;
    const int oh = (int)(t1 % OH);
    const long long bc = t1 / OH;

    const float sy = (float)oh * ((float)(H - 1) / (float)(OH - 1));
    const float sx = (float)ow * ((float)(W - 1) / (float)(OW - 1));
    const int y0 = (int)floorf(sy), x0 = (int)floorf(sx);
    const float fy = sy - (float)y0, fx = sx - (float)x0;
    const int y1 = min(y0 + 1, H - 1), x1 = min(x0 + 1, W - 1);
    const float* inp = In + bc * (long long)H * W;
    float v = (1.0f - fy) * (1.0f - fx) * inp[y0 * W + x0]
            + (1.0f - fy) * fx          * inp[y0 * W + x1]
            + fy          * (1.0f - fx) * inp[y1 * W + x0]
            + fy          * fx          * inp[y1 * W + x1];
    if (Res) v += Res[idx];
    Out[idx] = v;
}

// ---------------------------------------------------------------------------
// Host-side orchestration
// ---------------------------------------------------------------------------
static inline void run_conv3x3(const unsigned short* w, const float* x, float* out,
                               const float* bias, int O, int Cin,
                               int Bn, int H, int W, int flags, float scale,
                               hipStream_t s)
{
    const int S = Bn * H * W;
    dim3 grid((S + TILE_S - 1) / TILE_S, (O + TILE_O - 1) / TILE_O);
    gemm_conv_wmma<1><<<grid, 256, 0, s>>>(w, x, nullptr, out, bias,
                                           nullptr, nullptr, nullptr, nullptr,
                                           O, Cin * 9, Bn, Cin, H, W, flags, scale);
}

static inline void run_conv1x1(const unsigned short* w, const float* x, float* out,
                               const float* bias, int O, int Cin,
                               int Bn, int H, int W, int flags, float scale,
                               hipStream_t s)
{
    const int S = Bn * H * W;
    dim3 grid((S + TILE_S - 1) / TILE_S, (O + TILE_O - 1) / TILE_O);
    gemm_conv_wmma<2><<<grid, 256, 0, s>>>(w, x, nullptr, out, bias,
                                           nullptr, nullptr, nullptr, nullptr,
                                           O, Cin, Bn, Cin, H, W, flags, scale);
}

// One DCN block: offset-mask conv -> deformable gather -> WMMA einsum (+BN+ReLU)
// -> 2x upsample with residual add into outbuf (Bn, Cout, 2H, 2W).
static inline void run_dcn(const float* x, const float* const* P,
                           const unsigned short* wom_bf, const unsigned short* w_bf,
                           float* om, unsigned short* vmat, float* pre,
                           const float* res, float* outbuf,
                           int Bn, int Cin, int Cout, int H, int W,
                           hipStream_t s)
{
    // 1. offset/mask conv (C -> 27), bias only
    run_conv3x3(wom_bf, x, om, P[1], 27, Cin, Bn, H, W, /*bias*/1, 1.0f, s);
    // 2. deformable bilinear gather -> bf16 V matrix [Cin*9][Bn*H*W]
    {
        const int total = Bn * 9 * H * W;
        dcn_sample<<<(total + 255) / 256, 256, 0, s>>>(x, om, vmat, Bn, Cin, H, W);
    }
    // 3. einsum GEMM (plain bf16 matrix), fused bias + BN + ReLU
    {
        const int S = Bn * H * W;
        dim3 grid((S + TILE_S - 1) / TILE_S, (Cout + TILE_O - 1) / TILE_O);
        gemm_conv_wmma<0><<<grid, 256, 0, s>>>(w_bf, nullptr, vmat, pre, P[3],
                                               P[4], P[5], P[6], P[7],
                                               Cout, Cin * 9, Bn, 0, H, W,
                                               1 | 2 | 4, 1.0f);
    }
    // 4. 2x upsample + residual
    {
        const long long tot = (long long)Bn * Cout * 4 * H * W;
        upsample2x_add<<<(unsigned)((tot + 255) / 256), 256, 0, s>>>(pre, res, outbuf,
                                                                     Bn, Cout, H, W);
    }
}

extern "C" void kernel_launch(void* const* d_in, const int* in_sizes, int n_in,
                              void* d_out, int out_size, void* d_ws, size_t ws_size,
                              hipStream_t stream)
{
    (void)in_sizes; (void)n_in; (void)out_size; (void)ws_size;

    const float* P[63];
    for (int i = 0; i < 63; ++i) P[i] = (const float*)d_in[i];

    const float* x1 = P[0];  // (4,512,32,32)
    const float* x2 = P[1];  // (4,256,64,64)
    const float* x3 = P[2];  // (4,128,128,128)

    // -------- workspace carving --------
    char* wp = (char*)d_ws;
    auto carve = [&](size_t bytes) -> void* {
        void* p = (void*)wp;
        wp += (bytes + 255) & ~(size_t)255;
        return p;
    };
    unsigned short* vmat = (unsigned short*)carve((size_t)80 << 20);        // max 75.5MB
    float* pre  = (float*)carve((size_t)4 * 256 * 64 * 64 * 4);             // pre-upsample
    float* om   = (float*)carve((size_t)4 * 27 * 64 * 64 * 4);              // offset/mask
    float* x2p  = (float*)carve((size_t)4 * 256 * 64 * 64 * 4);
    float* x3p  = (float*)carve((size_t)4 * 128 * 128 * 128 * 4);
    float* xh   = (float*)carve((size_t)4 * 256 * 64 * 64 * 4);
    float* xw   = (float*)carve((size_t)4 * 256 * 64 * 64 * 4);
    float* hmx  = (float*)carve((size_t)4 * 128 * 128 * 128 * 4);
    float* whx  = (float*)carve((size_t)4 * 128 * 128 * 128 * 4);
    float* t1   = (float*)carve((size_t)4 * 128 * 128 * 128 * 4);
    float* t2   = (float*)carve((size_t)4 * 128 * 128 * 128 * 4);

    // -------- pre-convert all weight tensors to bf16 (O padded to 64 rows) ----
    struct WSpec { int idx; int O; int K; };
    static const WSpec ws18[18] = {
        {3, 27, 4608}, {5, 256, 4608},     // mdcn1:    w_om, w
        {11, 27, 2304}, {13, 128, 2304},   // mdcn2
        {19, 27, 4608}, {21, 256, 4608},   // mdcn1_hm
        {27, 27, 2304}, {29, 128, 2304},   // mdcn2_hm
        {35, 27, 4608}, {37, 256, 4608},   // mdcn1_wh
        {43, 27, 2304}, {45, 128, 2304},   // mdcn2_wh
        {51, 128, 1152}, {53, 128, 1152}, {55, 80, 128},   // hm head
        {57, 128, 1152}, {59, 64, 1152},  {61, 4, 64},     // wh head
    };
    const unsigned short* wb[18];
    for (int i = 0; i < 18; ++i) {
        const int Opad = (ws18[i].O + TILE_O - 1) / TILE_O * TILE_O;
        const int n    = ws18[i].O * ws18[i].K;
        unsigned short* dst = (unsigned short*)carve((size_t)Opad * ws18[i].K * 2);
        f32_to_bf16_kernel<<<(n + 255) / 256, 256, 0, stream>>>(P[ws18[i].idx], dst, n);
        wb[i] = dst;
    }

    // -------- trunk --------
    run_dcn(x1,  &P[3],  wb[0],  wb[1],  om, vmat, pre, x2,  x2p, 4, 512, 256, 32, 32, stream);
    run_dcn(x2p, &P[11], wb[2],  wb[3],  om, vmat, pre, x3,  x3p, 4, 256, 128, 64, 64, stream);

    // -------- hm branch --------
    run_dcn(x1,  &P[19], wb[4],  wb[5],  om, vmat, pre, x2p, xh,  4, 512, 256, 32, 32, stream);
    run_dcn(xh,  &P[27], wb[6],  wb[7],  om, vmat, pre, x3p, hmx, 4, 256, 128, 64, 64, stream);

    // -------- wh branch --------
    run_dcn(x1,  &P[35], wb[8],  wb[9],  om, vmat, pre, x2p, xw,  4, 512, 256, 32, 32, stream);
    run_dcn(xw,  &P[43], wb[10], wb[11], om, vmat, pre, x3p, whx, 4, 256, 128, 64, 64, stream);

    float* hm_out = (float*)d_out;                              // (4,80,128,128)
    float* wh_out = hm_out + (size_t)4 * 80 * 128 * 128;        // (4,4,128,128)

    // -------- hm head: conv3x3+relu, conv3x3+relu, conv1x1+bias --------
    run_conv3x3(wb[12], hmx, t1, P[52], 128, 128, 4, 128, 128, 1 | 4, 1.0f, stream);
    run_conv3x3(wb[13], t1,  t2, P[54], 128, 128, 4, 128, 128, 1 | 4, 1.0f, stream);
    run_conv1x1(wb[14], t2, hm_out, P[56], 80, 128, 4, 128, 128, 1, 1.0f, stream);

    // -------- wh head: conv3x3+relu, conv3x3+relu, conv1x1 -> relu -> *16 --------
    run_conv3x3(wb[15], whx, t1, P[58], 128, 128, 4, 128, 128, 1 | 4, 1.0f, stream);
    run_conv3x3(wb[16], t1,  t2, P[60], 64, 128, 4, 128, 128, 1 | 4, 1.0f, stream);
    run_conv1x1(wb[17], t2, wh_out, P[62], 4, 64, 4, 128, 128, 1 | 4 | 8, 16.0f, stream);
}
